// MultiHeadLatentAttention_81896436400179
// MI455X (gfx1250) — matrix operand order repaired and checked by reading, not compiled, not run on hardware
//
#include <hip/hip_runtime.h>
#include <hip/hip_bf16.h>
#include <math.h>
#include <stdint.h>

// ---------------- problem constants ----------------
#define BB    4
#define TT    2048
#define DIMM  512
#define HEADS 8
#define HD    64
#define RANK  256
#define WIN   256
#define BT    (BB * TT)          // 8192 rows

typedef __attribute__((ext_vector_type(16))) _Float16 v16h;
typedef __attribute__((ext_vector_type(8)))  float    v8f;

// ---------------- CDNA5 async copy (global -> LDS), ASYNCcnt tracked ----------------
__device__ inline void async_g2l_b128(void* lds, const void* gptr) {
  uint32_t loff = (uint32_t)(uintptr_t)lds;   // low 32b of generic shared ptr == LDS offset
  asm volatile("global_load_async_to_lds_b128 %0, %1, off"
               :: "v"(loff), "v"(gptr) : "memory");
}
template <int N>
__device__ inline void wait_async() {
  asm volatile("s_wait_asynccnt %0" :: "i"(N) : "memory");
}

// ---------------- WMMA helpers (wave32; A/B use contiguous k = 16h+s, not the ISA table order; C/D per ISA 7.12.2) ----------------
__device__ inline v8f wmma_f16(v16h a, v16h b, v8f c) {
  return __builtin_amdgcn_wmma_f32_16x16x32_f16(false, a, false, b, (short)0, c,
                                                false, false);
}

// A fragment 16x32 (MxK) from row-major f32 source: (m,k)=base[m*ld+k].
// lane<16: row=lane, K=0..15; lane>=16: row=lane-16, K=16..31
__device__ inline v16h load_a_frag_f32(const float* base, int ld, int lane) {
  int m    = lane & 15;
  int koff = (lane >> 4) << 4;
  const float* p = base + (size_t)m * ld + koff;
  float4 t0 = *(const float4*)(p);
  float4 t1 = *(const float4*)(p + 4);
  float4 t2 = *(const float4*)(p + 8);
  float4 t3 = *(const float4*)(p + 12);
  v16h f;
  f[0]=(_Float16)t0.x; f[1]=(_Float16)t0.y; f[2]=(_Float16)t0.z; f[3]=(_Float16)t0.w;
  f[4]=(_Float16)t1.x; f[5]=(_Float16)t1.y; f[6]=(_Float16)t1.z; f[7]=(_Float16)t1.w;
  f[8]=(_Float16)t2.x; f[9]=(_Float16)t2.y; f[10]=(_Float16)t2.z; f[11]=(_Float16)t2.w;
  f[12]=(_Float16)t3.x; f[13]=(_Float16)t3.y; f[14]=(_Float16)t3.z; f[15]=(_Float16)t3.w;
  return f;
}

// Same A fragment from an f16 tile (two 16B contiguous chunks -> 2x ds_load_b128).
__device__ inline v16h load_a_frag_h16(const _Float16* base, int ld, int lane) {
  int m    = lane & 15;
  int koff = (lane >> 4) << 4;
  const _Float16* p = base + m * ld + koff;
  v16h f;
#pragma unroll
  for (int i = 0; i < 8; ++i) f[i]     = p[i];
#pragma unroll
  for (int i = 0; i < 8; ++i) f[i + 8] = p[i + 8];
  return f;
}

// B fragment 32x16 (KxN) from f32 source where K is contiguous per n: (k,n)=base[n*ld+k].
// lane<16: n=lane, K=0..15; lane>=16: n=lane-16, K=16..31.
__device__ inline v16h load_b_frag_nk_f32(const float* base, int ld, int lane) {
  int n    = lane & 15;
  int koff = (lane >> 4) << 4;
  const float* p = base + (size_t)n * ld + koff;
  float4 t0 = *(const float4*)(p);
  float4 t1 = *(const float4*)(p + 4);
  float4 t2 = *(const float4*)(p + 8);
  float4 t3 = *(const float4*)(p + 12);
  v16h f;
  f[0]=(_Float16)t0.x; f[1]=(_Float16)t0.y; f[2]=(_Float16)t0.z; f[3]=(_Float16)t0.w;
  f[4]=(_Float16)t1.x; f[5]=(_Float16)t1.y; f[6]=(_Float16)t1.z; f[7]=(_Float16)t1.w;
  f[8]=(_Float16)t2.x; f[9]=(_Float16)t2.y; f[10]=(_Float16)t2.z; f[11]=(_Float16)t2.w;
  f[12]=(_Float16)t3.x; f[13]=(_Float16)t3.y; f[14]=(_Float16)t3.z; f[15]=(_Float16)t3.w;
  return f;
}

// B fragment 32x16 from f16 source, K contiguous per n: (k,n)=base[n*ld+k].
// 16 contiguous halfs per lane -> 2x ds_load_b128, zero VALU.
__device__ inline v16h load_b_frag_nk_h16(const _Float16* base, int ld, int lane) {
  int n    = lane & 15;
  int koff = (lane >> 4) << 4;
  const _Float16* p = base + n * ld + koff;
  v16h f;
#pragma unroll
  for (int i = 0; i < 16; ++i) f[i] = p[i];
  return f;
}

// C/D store: VGPR r -> row mbase+r (mbase=8 for lanes>=16), lane&15 -> col.
__device__ inline void store_c_frag(float* base, int ld, int lane, v8f c) {
  int n  = lane & 15;
  int mb = (lane >> 4) << 3;
#pragma unroll
  for (int r = 0; r < 8; ++r) base[(size_t)(mb + r) * ld + n] = c[r];
}

// ---------------- GEMM: Y[M,N] = X[M,K]*W[N,K]^T ----------------
// A tile 128x32 f32 via async-to-LDS (ld 36 -> conflict-free); B tile 64x32 staged
// as f16 once per block (ld 40 -> conflict-free, 16B-aligned fragment reads).
#define LDA 36
#define LDB 40
__global__ void __launch_bounds__(256)
gemm_xwT_kernel(const float* __restrict__ X, const float* __restrict__ W,
                float* __restrict__ Y, int N, int K) {
  __shared__ float    As[2][128 * LDA];
  __shared__ _Float16 Bs[2][64 * LDB];
  int tid  = threadIdx.x;
  int wave = tid >> 5, lane = tid & 31;
  int row0 = blockIdx.x * 128;
  int n0   = blockIdx.y * 64;
  int nk   = K >> 5;

  int br = tid >> 3, bc = (tid & 7) << 2;   // this thread's two B chunks: rows br, br+32

  auto stageA = [&](int buf, int k0) {
#pragma unroll
    for (int t = 0; t < 4; ++t) {
      int idx = tid + t * 256;
      int r = idx >> 3, c4 = (idx & 7) << 2;
      async_g2l_b128(&As[buf][r * LDA + c4],
                     X + (size_t)(row0 + r) * K + k0 + c4);
    }
  };
  auto loadB = [&](int k0, float4& t0, float4& t1) {
    t0 = *(const float4*)(W + (size_t)(n0 + br) * K + k0 + bc);
    t1 = *(const float4*)(W + (size_t)(n0 + br + 32) * K + k0 + bc);
  };
  auto storeB = [&](int buf, float4 t0, float4 t1) {
    _Float16* p0 = &Bs[buf][br * LDB + bc];
    p0[0]=(_Float16)t0.x; p0[1]=(_Float16)t0.y; p0[2]=(_Float16)t0.z; p0[3]=(_Float16)t0.w;
    _Float16* p1 = &Bs[buf][(br + 32) * LDB + bc];
    p1[0]=(_Float16)t1.x; p1[1]=(_Float16)t1.y; p1[2]=(_Float16)t1.z; p1[3]=(_Float16)t1.w;
  };

  v8f acc0 = {}, acc1 = {}, acc2 = {}, acc3 = {};
  // prologue: tile 0
  stageA(0, 0);
  { float4 t0, t1; loadB(0, t0, t1); storeB(0, t0, t1); }
  wait_async<0>();
  __syncthreads();

  for (int kt = 0; kt < nk; ++kt) {
    int cur = kt & 1;
    float4 t0, t1;
    bool hasNext = (kt + 1 < nk);
    if (hasNext) {
      stageA(cur ^ 1, (kt + 1) << 5);      // async A copy overlaps compute
      loadB((kt + 1) << 5, t0, t1);        // issue B loads; convert after compute
    }

    const float* Ab = &As[cur][(wave * 16) * LDA];
    v16h a  = load_a_frag_f32(Ab, LDA, lane);
    v16h b0 = load_b_frag_nk_h16(&Bs[cur][ 0 * LDB], LDB, lane);
    v16h b1 = load_b_frag_nk_h16(&Bs[cur][16 * LDB], LDB, lane);
    v16h b2 = load_b_frag_nk_h16(&Bs[cur][32 * LDB], LDB, lane);
    v16h b3 = load_b_frag_nk_h16(&Bs[cur][48 * LDB], LDB, lane);
    acc0 = wmma_f16(a, b0, acc0);
    acc1 = wmma_f16(a, b1, acc1);
    acc2 = wmma_f16(a, b2, acc2);
    acc3 = wmma_f16(a, b3, acc3);

    if (hasNext) storeB(cur ^ 1, t0, t1);  // cvt+ds_store after wmma issue
    wait_async<0>();
    __syncthreads();
  }

  float* ybase = Y + (size_t)(row0 + wave * 16) * N + n0;
  store_c_frag(ybase +  0, N, lane, acc0);
  store_c_frag(ybase + 16, N, lane, acc1);
  store_c_frag(ybase + 32, N, lane, acc2);
  store_c_frag(ybase + 48, N, lane, acc3);
}

// ---------------- RMSNorm over RANK=256 columns ----------------
__global__ void __launch_bounds__(256)
rmsnorm_kernel(float* __restrict__ C, const float* __restrict__ w) {
  __shared__ float red[256];
  int row = blockIdx.x;
  int tid = threadIdx.x;
  float v = C[(size_t)row * RANK + tid];
  red[tid] = v * v;
  __syncthreads();
#pragma unroll
  for (int s = 128; s > 0; s >>= 1) {
    if (tid < s) red[tid] += red[tid + s];
    __syncthreads();
  }
  float r = rsqrtf(red[0] * (1.0f / RANK) + 1e-6f);
  C[(size_t)row * RANK + tid] = v * r * w[tid];
}

// ---------------- RoPE on Q and K-half of KV ----------------
__global__ void __launch_bounds__(256)
rope_kernel(float* __restrict__ Q, float* __restrict__ KV) {
  int row = blockIdx.x;                  // b*T + t
  int t = row & (TT - 1);
  int h = threadIdx.x >> 5;
  int d = threadIdx.x & 31;
  float ang = (float)t * __expf(-(float)d * (9.210340371976184f / 32.0f));
  float s, c;
  __sincosf(ang, &s, &c);
  size_t qb = (size_t)row * DIMM + h * HD + d;
  float q1 = Q[qb], q2 = Q[qb + 32];
  Q[qb]      = q1 * c - q2 * s;
  Q[qb + 32] = q2 * c + q1 * s;
  size_t kb = (size_t)row * (2 * DIMM) + h * HD + d;
  float k1 = KV[kb], k2 = KV[kb + 32];
  KV[kb]      = k1 * c - k2 * s;
  KV[kb + 32] = k2 * c + k1 * s;
}

// ---------------- banded flash attention ----------------
// One wave per (b, h, 16-query tile). 4 waves/block (LDS ~58KB).
// K tiles: double-buffered async-to-LDS (f32, ld 68). V tiles: reg-staged
// transposed f16 vT[d][k] (ld 40), K rows 16..31 pre-zeroed = WMMA K=32 pad.
#define LDK 68
#define LDV 40
__global__ void __launch_bounds__(128)
banded_attn_kernel(const float* __restrict__ Q, const float* __restrict__ KV,
                   float* __restrict__ O) {
  __shared__ float    kbuf[4][2][16 * LDK];
  __shared__ _Float16 vT[4][64 * LDV];
  __shared__ _Float16 lds_p[4][16 * 32];

  int tid  = threadIdx.x;
  int wave = tid >> 5;
  int lane = tid & 31;

  // zero pads once: P cols 16..31, vT k rows 16..39
  for (int idx = lane; idx < 16 * 16; idx += 32) {
    int m = idx >> 4;
    lds_p[wave][m * 32 + 16 + (idx & 15)] = (_Float16)0.0f;
  }
  for (int idx = lane; idx < 64 * 24; idx += 32) {
    int d = idx / 24, kk = 16 + idx % 24;
    vT[wave][d * LDV + kk] = (_Float16)0.0f;
  }
  __builtin_amdgcn_wave_barrier();

  int tile = blockIdx.x * 4 + wave;
  int i0 = (tile & (TT / 16 - 1)) * 16;
  int bh = tile >> 7;                      // T/16 == 128
  int h = bh & (HEADS - 1);
  int b = bh >> 3;

  const float scale = 0.125f;              // 64^-0.5
  int mb = (lane >> 4) << 3;
  int nn = lane & 15;
  int crr = lane >> 1;                     // V staging: this lane's (row, col4)... see below

  const float* qbase = Q + (size_t)(b * TT + i0) * DIMM + h * HD;
  v16h aq0 = load_a_frag_f32(qbase +  0, DIMM, lane);
  v16h aq1 = load_a_frag_f32(qbase + 32, DIMM, lane);

  float mrow[8], lrow[8];
  v8f oacc0 = {}, oacc1 = {}, oacc2 = {}, oacc3 = {};
#pragma unroll
  for (int r = 0; r < 8; ++r) { mrow[r] = -3.0e38f; lrow[r] = 0.0f; }

  int lo = i0 - WIN; if (lo < 0) lo = 0;
  int hi = i0 + 16 + WIN; if (hi > TT) hi = TT;
  int nIter = (hi - lo) >> 4;

  auto stageK = [&](int buf, int j0) {     // 16x64 f32, 8 async b128 per lane
#pragma unroll
    for (int t = 0; t < 8; ++t) {
      int idx = lane + t * 32;
      int rr = idx >> 4, c4 = (idx & 15) << 2;
      async_g2l_b128(&kbuf[wave][buf][rr * LDK + c4],
                     KV + (size_t)(b * TT + j0 + rr) * (2 * DIMM) + h * HD + c4);
    }
  };

  stageK(0, lo);                           // prologue

  for (int kt = 0; kt < nIter; ++kt) {
    int j0 = lo + (kt << 4);
    bool hasNext = (kt + 1 < nIter);
    int cur = kt & 1;

    // V tile: global f32 -> regs; stored transposed f16 after scores
    float4 vt[8];
#pragma unroll
    for (int t = 0; t < 8; ++t) {
      int idx = lane + t * 32;
      int rr = idx >> 4, c4 = (idx & 15) << 2;
      vt[t] = *(const float4*)(KV + (size_t)(b * TT + j0 + rr) * (2 * DIMM)
                               + DIMM + h * HD + c4);
    }
    if (hasNext) stageK(cur ^ 1, j0 + 16); // async K(j+1) overlaps everything below

    // wait for K(j): with K(j+1) outstanding (8 ops, in-order) threshold 8, else 0
    if (hasNext) wait_async<8>(); else wait_async<0>();
    __builtin_amdgcn_wave_barrier();

    // ---- scores S = Q K^T ----
    const float* kb = &kbuf[wave][cur][0];
    v16h bk0 = load_b_frag_nk_f32(kb +  0, LDK, lane);
    v16h bk1 = load_b_frag_nk_f32(kb + 32, LDK, lane);
    v8f s = {};
    s = wmma_f16(aq0, bk0, s);
    s = wmma_f16(aq1, bk1, s);

    // store V transposed as f16 (reads of previous tile finished before here)
#pragma unroll
    for (int t = 0; t < 8; ++t) {
      int idx = lane + t * 32;
      int rr = idx >> 4, c4 = (idx & 15) << 2;
      vT[wave][(c4 + 0) * LDV + rr] = (_Float16)vt[t].x;
      vT[wave][(c4 + 1) * LDV + rr] = (_Float16)vt[t].y;
      vT[wave][(c4 + 2) * LDV + rr] = (_Float16)vt[t].z;
      vT[wave][(c4 + 3) * LDV + rr] = (_Float16)vt[t].w;
    }

    // ---- online softmax ----
    float alpha[8];
#pragma unroll
    for (int r = 0; r < 8; ++r) {
      int i = i0 + mb + r;
      int j = j0 + nn;
      float sv = s[r] * scale;
      int diff = j - i; if (diff < 0) diff = -diff;
      if (diff > WIN) sv = -3.0e38f;
      float red = sv;
      red = fmaxf(red, __shfl_xor(red, 1, 32));
      red = fmaxf(red, __shfl_xor(red, 2, 32));
      red = fmaxf(red, __shfl_xor(red, 4, 32));
      red = fmaxf(red, __shfl_xor(red, 8, 32));
      float mnew = fmaxf(mrow[r], red);
      alpha[r] = __expf(mrow[r] - mnew);
      mrow[r] = mnew;
      float p = __expf(sv - mnew);
      float rs = p;
      rs += __shfl_xor(rs, 1, 32);
      rs += __shfl_xor(rs, 2, 32);
      rs += __shfl_xor(rs, 4, 32);
      rs += __shfl_xor(rs, 8, 32);
      lrow[r] = lrow[r] * alpha[r] + rs;
      lds_p[wave][(mb + r) * 32 + nn] = (_Float16)p;
    }
    __builtin_amdgcn_wave_barrier();

    // ---- O = O*alpha + P V ----
    v16h pa  = load_a_frag_h16(&lds_p[wave][0], 32, lane);
    const _Float16* vb = &vT[wave][0];
    v16h vb0 = load_b_frag_nk_h16(vb +  0 * LDV, LDV, lane);
    v16h vb1 = load_b_frag_nk_h16(vb + 16 * LDV, LDV, lane);
    v16h vb2 = load_b_frag_nk_h16(vb + 32 * LDV, LDV, lane);
    v16h vb3 = load_b_frag_nk_h16(vb + 48 * LDV, LDV, lane);
#pragma unroll
    for (int r = 0; r < 8; ++r) {
      oacc0[r] *= alpha[r]; oacc1[r] *= alpha[r];
      oacc2[r] *= alpha[r]; oacc3[r] *= alpha[r];
    }
    oacc0 = wmma_f16(pa, vb0, oacc0);
    oacc1 = wmma_f16(pa, vb1, oacc1);
    oacc2 = wmma_f16(pa, vb2, oacc2);
    oacc3 = wmma_f16(pa, vb3, oacc3);
    __builtin_amdgcn_wave_barrier();
  }

  float* obase = O + (size_t)(b * TT + i0) * DIMM + h * HD;
#pragma unroll
  for (int r = 0; r < 8; ++r) {
    float inv = 1.0f / lrow[r];
    obase[(size_t)(mb + r) * DIMM +  0 + nn] = oacc0[r] * inv;
    obase[(size_t)(mb + r) * DIMM + 16 + nn] = oacc1[r] * inv;
    obase[(size_t)(mb + r) * DIMM + 32 + nn] = oacc2[r] * inv;
    obase[(size_t)(mb + r) * DIMM + 48 + nn] = oacc3[r] * inv;
  }
}

// ---------------- launch ----------------
extern "C" void kernel_launch(void* const* d_in, const int* in_sizes, int n_in,
                              void* d_out, int out_size, void* d_ws, size_t ws_size,
                              hipStream_t stream) {
  const float* x         = (const float*)d_in[0];
  const float* Wq        = (const float*)d_in[1];
  const float* Wkv_down  = (const float*)d_in[2];
  const float* kv_norm_w = (const float*)d_in[3];
  const float* Wkv_up    = (const float*)d_in[4];
  const float* Wout      = (const float*)d_in[5];
  float* out = (float*)d_out;

  char* ws = (char*)d_ws;
  float* Q    = (float*)(ws);                                   // 16 MB [BT,512]
  float* CKV  = (float*)(ws + (size_t)16 * 1024 * 1024);        //  8 MB [BT,256]
  float* KV   = (float*)(ws + (size_t)24 * 1024 * 1024);        // 32 MB [BT,1024]
  float* ATTN = (float*)(ws + (size_t)56 * 1024 * 1024);        // 16 MB [BT,512]

  dim3 blk(256);
  gemm_xwT_kernel<<<dim3(BT / 128, DIMM / 64), blk, 0, stream>>>(x, Wq, Q, DIMM, DIMM);
  gemm_xwT_kernel<<<dim3(BT / 128, RANK / 64), blk, 0, stream>>>(x, Wkv_down, CKV, RANK, DIMM);
  rmsnorm_kernel<<<dim3(BT), blk, 0, stream>>>(CKV, kv_norm_w);
  gemm_xwT_kernel<<<dim3(BT / 128, (2 * DIMM) / 64), blk, 0, stream>>>(CKV, Wkv_up, KV, 2 * DIMM, RANK);
  rope_kernel<<<dim3(BT), blk, 0, stream>>>(Q, KV);
  banded_attn_kernel<<<dim3(BB * HEADS * (TT / 16) / 4), dim3(128), 0, stream>>>(Q, KV, ATTN);
  gemm_xwT_kernel<<<dim3(BT / 128, DIMM / 64), blk, 0, stream>>>(ATTN, Wout, out, DIMM, DIMM);
}
